// FasterRCNNSEMob_90563680404093
// MI455X (gfx1250) — matrix operand order repaired and checked by reading, not compiled, not run on hardware
//
#include <hip/hip_runtime.h>

typedef __attribute__((ext_vector_type(16))) _Float16 v16h;
typedef __attribute__((ext_vector_type(8)))  _Float16 v8h;
typedef __attribute__((ext_vector_type(8)))  float    v8f;
typedef __attribute__((ext_vector_type(4)))  float    v4f;

// ---- WMMA fragment layout as used here (wave32; A/B in contiguous k, not the ISA table order) ----
// A (16x32 f16): lane L<16 -> M=L, K=k0..k0+15;
//                lane L>=16 -> M=L-16, K=k0+16..k0+31.
// B (32x16 f16): lane L<16 -> N=L, K=k0..k0+15; lane L>=16 -> N=L-16, K=k0+16..k0+31.
// C/D: VGPR r: lanes 0-15 -> (M=r, N=lane); lanes 16-31 -> (M=r+8, N=lane-16).

__device__ __forceinline__ v16h cat8(v8h a0, v8h a1) {
  return __builtin_shufflevector(a0, a1, 0, 1, 2, 3, 4, 5, 6, 7, 8, 9, 10, 11, 12, 13, 14, 15);
}
__device__ __forceinline__ v16h cvt16(v4f b0, v4f b1, v4f b2, v4f b3) {
  v16h b;
#pragma unroll
  for (int i = 0; i < 4; ++i) {
    b[i]      = (_Float16)b0[i];
    b[4 + i]  = (_Float16)b1[i];
    b[8 + i]  = (_Float16)b2[i];
    b[12 + i] = (_Float16)b3[i];
  }
  return b;
}

// ---------------- helpers ----------------
__device__ __forceinline__ unsigned enc_f(float f) {
  unsigned u = __float_as_uint(f);
  return (u & 0x80000000u) ? ~u : (u | 0x80000000u);
}
__device__ __forceinline__ float dec_f(unsigned u) {
  unsigned v = (u & 0x80000000u) ? (u & 0x7FFFFFFFu) : ~u;
  return __uint_as_float(v);
}
__device__ __forceinline__ float iou_one(float ax1, float ay1, float ax2, float ay2,
                                         float bx1, float by1, float bx2, float by2) {
  float tlx = fmaxf(ax1, bx1), tly = fmaxf(ay1, by1);
  float brx = fminf(ax2, bx2), bry = fminf(ay2, by2);
  float w = fmaxf(brx - tlx, 0.f), h = fmaxf(bry - tly, 0.f);
  float inter = w * h;
  float aa = (ax2 - ax1) * (ay2 - ay1);
  float ab = (bx2 - bx1) * (by2 - by1);
  return inter / (aa + ab - inter + 1e-9f);
}
__device__ __forceinline__ float smooth_term(float x) {
  float s = (x < 0.5f) ? 0.5f * x * x : 0.f;
  if (x > 0.5f) s += x - 0.5f;
  return s;
}

// ---------------- im2col (3x3, pad=1) : feat [512][50][50] f32 -> A [2512][4608] f16 ----------------
__global__ void im2col_k(const float* __restrict__ feat, _Float16* __restrict__ A) {
  int i = blockIdx.x * blockDim.x + threadIdx.x;
  if (i >= 2512 * 4608) return;
  int m = i / 4608, k = i - m * 4608;
  float v = 0.f;
  if (m < 2500) {
    int y = m / 50, x = m - y * 50;
    int c = k / 9, rem = k - c * 9;
    int rr = rem / 3, ss = rem - rr * 3;
    int yy = y + rr - 1, xx = x + ss - 1;
    if (yy >= 0 && yy < 50 && xx >= 0 && xx < 50) v = feat[c * 2500 + yy * 50 + xx];
  }
  A[i] = (_Float16)v;
}

// ---------------- generic one-wave 16x16-tile GEMM ----------------
// A: f16 [>=gridDim.x*16][lda] (padded rows; garbage rows discarded by store guard)
// btrans=1: B f32 [N][ldb] (N-major)  -> vectorized b128 loads
// btrans=0: B f32 [K][ldb] (K-major)  -> lane-coalesced dword streams
__global__ void gemm_tile(const _Float16* __restrict__ A, int lda,
                          const float* __restrict__ B, int ldb, int btrans,
                          const float* __restrict__ bias,
                          float* __restrict__ Cf, _Float16* __restrict__ Ch,
                          int ldc, int M, int N, int K, int relu) {
  int lane = threadIdx.x & 31;
  int lm = lane & 15;
  int hi = (lane >> 4) & 1;
  int mt = blockIdx.x * 16, nt = blockIdx.y * 16;
  int m = mt + lm;
  int n = nt + lm;
  int nb = (n < N) ? n : 0;  // clamp: clamped columns discarded at store
  const _Float16* Arow = A + (size_t)m * lda + hi * 16;
  v8f acc = {0.f, 0.f, 0.f, 0.f, 0.f, 0.f, 0.f, 0.f};
  if (btrans) {
    const float* Brow = B + (size_t)nb * ldb + hi * 16;
    for (int k0 = 0; k0 < K; k0 += 32) {
      v8h a0 = *(const v8h*)(Arow + k0);
      v8h a1 = *(const v8h*)(Arow + k0 + 8);
      v4f b0 = *(const v4f*)(Brow + k0);
      v4f b1 = *(const v4f*)(Brow + k0 + 4);
      v4f b2 = *(const v4f*)(Brow + k0 + 8);
      v4f b3 = *(const v4f*)(Brow + k0 + 12);
      acc = __builtin_amdgcn_wmma_f32_16x16x32_f16(false, cat8(a0, a1), false,
                                                   cvt16(b0, b1, b2, b3), (short)0, acc,
                                                   false, false);
    }
  } else {
    const float* Bcol = B + (size_t)hi * 16 * ldb + nb;
    for (int k0 = 0; k0 < K; k0 += 32) {
      v8h a0 = *(const v8h*)(Arow + k0);
      v8h a1 = *(const v8h*)(Arow + k0 + 8);
      v16h b;
#pragma unroll
      for (int h = 0; h < 16; ++h) b[h] = (_Float16)Bcol[(size_t)(k0 + h) * ldb];
      acc = __builtin_amdgcn_wmma_f32_16x16x32_f16(false, cat8(a0, a1), false, b, (short)0, acc,
                                                   false, false);
    }
  }
  float bi = bias[nb];
  int mrow = mt + hi * 8;
#pragma unroll
  for (int r = 0; r < 8; ++r) {
    int mm = mrow + r;
    if (mm < M && n < N) {
      float v = acc[r] + bi;
      if (relu) v = fmaxf(v, 0.f);
      if (Ch) Ch[(size_t)mm * ldc + n] = (_Float16)v;
      else    Cf[(size_t)mm * ldc + n] = v;
    }
  }
}

// ---------------- M=128 FC GEMM: each wave owns a 128x16 strip; B (K-major f32) streamed once ----------------
__global__ void gemm_m128_fc(const _Float16* __restrict__ A, int lda,
                             const float* __restrict__ B, int ldb,
                             const float* __restrict__ bias,
                             _Float16* __restrict__ Ch, int ldc, int K) {
  int lane = threadIdx.x & 31;
  int wave = threadIdx.x >> 5;
  int lm = lane & 15;
  int hi = (lane >> 4) & 1;
  int nt = blockIdx.x * 128 + wave * 16;
  int n = nt + lm;
  const float* Bcol = B + (size_t)hi * 16 * ldb + n;
  const _Float16* Abase = A + (size_t)lm * lda + hi * 16;
  v8f z = {0.f, 0.f, 0.f, 0.f, 0.f, 0.f, 0.f, 0.f};
  v8f acc[8];
#pragma unroll
  for (int i = 0; i < 8; ++i) acc[i] = z;
  for (int k0 = 0; k0 < K; k0 += 32) {
    v16h b;
#pragma unroll
    for (int h = 0; h < 16; ++h) b[h] = (_Float16)Bcol[(size_t)(k0 + h) * ldb];
    if (k0 + 32 < K) __builtin_prefetch(&Bcol[(size_t)(k0 + 32) * ldb], 0, 1);
#pragma unroll
    for (int mt = 0; mt < 8; ++mt) {
      const _Float16* Arow = Abase + (size_t)(mt * 16) * lda + k0;
      v8h a0 = *(const v8h*)(Arow);
      v8h a1 = *(const v8h*)(Arow + 8);
      acc[mt] = __builtin_amdgcn_wmma_f32_16x16x32_f16(false, cat8(a0, a1), false, b, (short)0,
                                                       acc[mt], false, false);
    }
  }
  float bi = bias[n];
#pragma unroll
  for (int mt = 0; mt < 8; ++mt) {
    int mrow = mt * 16 + hi * 8;
#pragma unroll
    for (int r = 0; r < 8; ++r)
      Ch[(size_t)(mrow + r) * ldc + n] = (_Float16)fmaxf(acc[mt][r] + bi, 0.f);
  }
}

// ---------------- RoI max pool (7x7, 2x2 samples) -> f16 [128][25088] ----------------
__global__ void roipool_k(const float* __restrict__ feat, const float* __restrict__ rois,
                          _Float16* __restrict__ pool) {
  int i = blockIdx.x * blockDim.x + threadIdx.x;
  if (i >= 128 * 512 * 49) return;
  int r = i / (512 * 49);
  int rem = i % (512 * 49);
  int c = rem / 49;
  int pq = rem % 49;
  int py = pq / 7, px = pq % 7;
  const float s = 1.0f / 16.0f;
  float x1 = rois[r * 4 + 0] * s, y1 = rois[r * 4 + 1] * s;
  float x2 = rois[r * 4 + 2] * s, y2 = rois[r * 4 + 3] * s;
  float maxv = -3.4e38f;
#pragma unroll
  for (int sy = 0; sy < 2; ++sy) {
    float fy = ((float)py + (sy + 0.5f) * 0.5f) / 7.0f;
    int iy = (int)fminf(fmaxf(floorf(y1 + fy * (y2 - y1)), 0.f), 49.f);
#pragma unroll
    for (int sx = 0; sx < 2; ++sx) {
      float fx = ((float)px + (sx + 0.5f) * 0.5f) / 7.0f;
      int ix = (int)fminf(fmaxf(floorf(x1 + fx * (x2 - x1)), 0.f), 49.f);
      maxv = fmaxf(maxv, feat[c * 2500 + iy * 50 + ix]);
    }
  }
  pool[(size_t)r * 25088 + c * 49 + pq] = (_Float16)maxv;
}

// ---------------- anchor target pass 1: row max/argmax + per-GT column max ----------------
__global__ void anchor1_k(const float* __restrict__ anchors, const float* __restrict__ bb,
                          float* __restrict__ ioumax, int* __restrict__ amax,
                          unsigned* __restrict__ gtenc) {
  int a = blockIdx.x * blockDim.x + threadIdx.x;
  if (a >= 22500) return;
  float ax1 = anchors[a * 4 + 0], ay1 = anchors[a * 4 + 1];
  float ax2 = anchors[a * 4 + 2], ay2 = anchors[a * 4 + 3];
  bool valid = (ax1 >= 0.f) && (ay1 >= 0.f) && (ax2 <= 800.f) && (ay2 <= 800.f);
  float best = -2.f;
  int bi = 0;
  for (int g = 0; g < 16; ++g) {
    float v = iou_one(ax1, ay1, ax2, ay2, bb[g * 4], bb[g * 4 + 1], bb[g * 4 + 2], bb[g * 4 + 3]);
    float vm = valid ? v : -1.f;
    if (vm > best) { best = vm; bi = g; }
    atomicMax(&gtenc[g], enc_f(vm));
  }
  ioumax[a] = best;
  amax[a] = bi;
}

// ---------------- anchor target pass 2: labels + regression targets ----------------
__global__ void anchor2_k(const float* __restrict__ anchors, const float* __restrict__ bb,
                          const float* __restrict__ ioumax, const int* __restrict__ amax,
                          const unsigned* __restrict__ gtenc,
                          int* __restrict__ tcls, float* __restrict__ tloc) {
  int a = blockIdx.x * blockDim.x + threadIdx.x;
  if (a >= 22500) return;
  float ax1 = anchors[a * 4 + 0], ay1 = anchors[a * 4 + 1];
  float ax2 = anchors[a * 4 + 2], ay2 = anchors[a * 4 + 3];
  bool valid = (ax1 >= 0.f) && (ay1 >= 0.f) && (ax2 <= 800.f) && (ay2 <= 800.f);
  float best = ioumax[a];
  int tc = -1;
  if (valid) {
    if (best < 0.3f) tc = 0;
    if (best >= 0.7f) tc = 1;
  }
  bool match = false;
  for (int g = 0; g < 16; ++g) {
    float v = iou_one(ax1, ay1, ax2, ay2, bb[g * 4], bb[g * 4 + 1], bb[g * 4 + 2], bb[g * 4 + 3]);
    float vm = valid ? v : -1.f;
    if (vm == dec_f(gtenc[g])) match = true;
  }
  if (match && valid) tc = 1;
  tcls[a] = tc;
  float o0 = 0.f, o1 = 0.f, o2 = 0.f, o3 = 0.f;
  if (valid) {
    int g = amax[a];
    float gx1 = bb[g * 4], gy1 = bb[g * 4 + 1], gx2 = bb[g * 4 + 2], gy2 = bb[g * 4 + 3];
    float aw = ax2 - ax1, ah = ay2 - ay1;
    float ax = ax1 + aw * 0.5f, ay = ay1 + ah * 0.5f;
    float gw = gx2 - gx1, gh = gy2 - gy1;
    float gx = gx1 + gw * 0.5f, gy = gy1 + gh * 0.5f;
    o0 = (gx - ax) / aw;
    o1 = (gy - ay) / ah;
    o2 = logf(gw / aw + 1e-9f);
    o3 = logf(gh / ah + 1e-9f);
  }
  tloc[a * 4 + 0] = o0; tloc[a * 4 + 1] = o1;
  tloc[a * 4 + 2] = o2; tloc[a * 4 + 3] = o3;
}

// ---------------- RPN losses ----------------
__global__ void rpn_loss_k(const float* __restrict__ cls, const float* __restrict__ loc,
                           const int* __restrict__ tcls, const float* __restrict__ tloc,
                           float* __restrict__ acc) {
  int a = blockIdx.x * blockDim.x + threadIdx.x;
  if (a >= 22500) return;
  int lab = tcls[a];
  if (lab >= 0) {
    float l0 = cls[a * 2], l1 = cls[a * 2 + 1];
    float mx = fmaxf(l0, l1);
    float lse = mx + logf(expf(l0 - mx) + expf(l1 - mx));
    float nll = lse - (lab ? l1 : l0);
    atomicAdd(&acc[0], nll);
    atomicAdd(&acc[1], 1.f);
  }
  if (lab > 0) {
    float s = 0.f;
#pragma unroll
    for (int c = 0; c < 4; ++c) {
      float x = fabsf(tloc[a * 4 + c] - loc[a * 4 + c]);
      s += smooth_term(x);
    }
    atomicAdd(&acc[2], s);
    atomicAdd(&acc[3], 1.f);
  }
}

// ---------------- FastRCNN losses ----------------
__global__ void roi_loss_k(const float* __restrict__ rcls, const float* __restrict__ rloc,
                           const int* __restrict__ labels, const float* __restrict__ gtloc,
                           float* __restrict__ acc) {
  int r = threadIdx.x;
  if (r >= 128) return;
  int lab = labels[r];
  float mx = -3.4e38f;
  for (int j = 0; j < 21; ++j) mx = fmaxf(mx, rcls[r * 21 + j]);
  float se = 0.f;
  for (int j = 0; j < 21; ++j) se += expf(rcls[r * 21 + j] - mx);
  float nll = mx + logf(se) - rcls[r * 21 + lab];
  atomicAdd(&acc[4], nll);
  if (lab > 0) {
    float s = 0.f;
#pragma unroll
    for (int c = 0; c < 4; ++c) {
      float x = fabsf(rloc[r * 84 + lab * 4 + c] - gtloc[r * 4 + c]);
      s += smooth_term(x);
    }
    atomicAdd(&acc[5], s);
    atomicAdd(&acc[6], 1.f);
  }
}

__global__ void final_k(const float* __restrict__ acc, float* __restrict__ out) {
  if (blockIdx.x == 0 && threadIdx.x == 0) {
    float rpn_cls = acc[0] / fmaxf(acc[1], 1.f);
    float rpn_loc = acc[2];
    float Nreg = fmaxf(acc[3], 1.f);
    float roi_cls = acc[4] / 128.f;
    float roi_loc = acc[5];
    float N2 = fmaxf(acc[6], 1.f);
    float t = roi_cls + 10.f / N2 * roi_loc + rpn_cls + 10.f / Nreg * rpn_loc;
    out[0] = rpn_cls; out[1] = rpn_loc; out[2] = roi_cls; out[3] = roi_loc; out[4] = t;
  }
}

// ---------------- host ----------------
extern "C" void kernel_launch(void* const* d_in, const int* in_sizes, int n_in,
                              void* d_out, int out_size, void* d_ws, size_t ws_size,
                              hipStream_t stream) {
  (void)in_sizes; (void)n_in; (void)out_size; (void)ws_size;
  const float* feat   = (const float*)d_in[0];
  const float* bboxes = (const float*)d_in[1];
  const float* rois   = (const float*)d_in[2];
  const float* gtloc  = (const float*)d_in[3];
  const float* anchors= (const float*)d_in[4];
  const float* rpn_w  = (const float*)d_in[5];
  const float* rpn_b  = (const float*)d_in[6];
  const float* reg_w  = (const float*)d_in[7];
  const float* reg_b  = (const float*)d_in[8];
  const float* cls_w  = (const float*)d_in[9];
  const float* cls_b  = (const float*)d_in[10];
  const float* fc1_w  = (const float*)d_in[11];
  const float* fc1_b  = (const float*)d_in[12];
  const float* fc2_w  = (const float*)d_in[13];
  const float* fc2_b  = (const float*)d_in[14];
  const float* hr_w   = (const float*)d_in[15];
  const float* hr_b   = (const float*)d_in[16];
  const float* hc_w   = (const float*)d_in[17];
  const float* hc_b   = (const float*)d_in[18];
  const int*   labels = (const int*)d_in[19];

  char* ws = (char*)d_ws;
  size_t off = 0;
  auto alloc = [&](size_t bytes) -> char* {
    char* p = ws + off;
    off = (off + bytes + 255) & ~(size_t)255;
    return p;
  };
  _Float16* col  = (_Float16*)alloc((size_t)2512 * 4608 * 2);  // im2col A
  _Float16* xt   = (_Float16*)alloc((size_t)2512 * 512 * 2);   // conv out, transposed
  float* loc     = (float*)alloc((size_t)22500 * 4 * 4);
  float* clsb    = (float*)alloc((size_t)22500 * 2 * 4);
  _Float16* pool = (_Float16*)alloc((size_t)128 * 25088 * 2);
  _Float16* h1   = (_Float16*)alloc((size_t)128 * 4096 * 2);
  _Float16* h2   = (_Float16*)alloc((size_t)128 * 4096 * 2);
  float* rloc    = (float*)alloc((size_t)128 * 84 * 4);
  float* rcls    = (float*)alloc((size_t)128 * 21 * 4);
  int* tcls      = (int*)alloc((size_t)22500 * 4);
  float* tloc    = (float*)alloc((size_t)22500 * 4 * 4);
  float* ioumax  = (float*)alloc((size_t)22500 * 4);
  int* amax      = (int*)alloc((size_t)22500 * 4);
  unsigned* gtenc= (unsigned*)alloc(64);
  float* acc     = (float*)alloc(64);

  hipMemsetAsync(gtenc, 0, 64, stream);  // 0 == encoded -NaN: identity for ordered max
  hipMemsetAsync(acc, 0, 64, stream);

  // RPN trunk + heads (conv3x3 as GEMM over im2col; 1x1 heads as GEMM over xt)
  im2col_k<<<((2512 * 4608) + 255) / 256, 256, 0, stream>>>(feat, col);
  gemm_tile<<<dim3(157, 32), 32, 0, stream>>>(col, 4608, rpn_w, 4608, 1, rpn_b,
                                              nullptr, xt, 512, 2500, 512, 4608, 1);
  gemm_tile<<<dim3(157, 3), 32, 0, stream>>>(xt, 512, reg_w, 512, 1, reg_b,
                                             loc, nullptr, 36, 2500, 36, 512, 0);
  gemm_tile<<<dim3(157, 2), 32, 0, stream>>>(xt, 512, cls_w, 512, 1, cls_b,
                                             clsb, nullptr, 18, 2500, 18, 512, 0);

  // FastRCNN head
  roipool_k<<<((128 * 512 * 49) + 255) / 256, 256, 0, stream>>>(feat, rois, pool);
  gemm_m128_fc<<<32, 256, 0, stream>>>(pool, 25088, fc1_w, 4096, fc1_b, h1, 4096, 25088);
  gemm_m128_fc<<<32, 256, 0, stream>>>(h1, 4096, fc2_w, 4096, fc2_b, h2, 4096, 4096);
  gemm_tile<<<dim3(8, 6), 32, 0, stream>>>(h2, 4096, hr_w, 84, 0, hr_b,
                                           rloc, nullptr, 84, 128, 84, 4096, 0);
  gemm_tile<<<dim3(8, 2), 32, 0, stream>>>(h2, 4096, hc_w, 21, 0, hc_b,
                                           rcls, nullptr, 21, 128, 21, 4096, 0);

  // Anchor targets + losses
  anchor1_k<<<(22500 + 255) / 256, 256, 0, stream>>>(anchors, bboxes, ioumax, amax, gtenc);
  anchor2_k<<<(22500 + 255) / 256, 256, 0, stream>>>(anchors, bboxes, ioumax, amax, gtenc, tcls, tloc);
  rpn_loss_k<<<(22500 + 255) / 256, 256, 0, stream>>>(clsb, loc, tcls, tloc, acc);
  roi_loss_k<<<1, 128, 0, stream>>>(rcls, rloc, labels, gtloc, acc);
  final_k<<<1, 32, 0, stream>>>(acc, (float*)d_out);
}
